// Spec_Encoder_68556267978855
// MI455X (gfx1250) — compile-verified, matrix operand order checked by reading, not run on hardware
//
#include <hip/hip_runtime.h>
#include <math.h>

typedef __attribute__((ext_vector_type(16))) _Float16 v16h;
typedef __attribute__((ext_vector_type(8)))  float    v8f;
typedef __attribute__((ext_vector_type(4)))  float    v4f;

// ---------------- problem constants ----------------
constexpr int BATCH = 512;
constexpr int FEAT  = 300;           // x is (512, 300), padded to 304
constexpr int NPOS  = 361;           // 19*19 grid positions
constexpr int NTILE = 23;            // ceil(361/16)
constexpr float EPS = 1e-5f;

constexpr int W1N = 256 * 512;       // 131072
constexpr int W2N = 128 * 256;       // 32768
constexpr int W3N = 64 * 128;        // 8192
constexpr int W4N = 4 * 64;          // 256
constexpr int W234N = W2N + W3N + W4N; // 41216 halfs (contiguous in ws)

// ---------------- LDS layout (bytes) ----------------
constexpr int H1_STRIDE = 264;       // 256 + pad (halfs)
constexpr int H2_STRIDE = 136;       // 128 + pad (halfs)
constexpr int M3_STRIDE = 72;        // 64 + pad  (halfs)

constexpr int OFF_XP  = 0;                       // 304 f32    = 1216 B
constexpr int OFF_H1  = OFF_XP + 1216;           // 8*16*264 h = 67584 B
constexpr int OFF_M3  = OFF_H1 + 67584;          // 8*16*72 h  = 18432 B
constexpr int OFF_H2  = OFF_M3 + 18432;          // 368*136 h  = 100096 B
constexpr int OFF_M4  = OFF_H2 + 100096;         // 368*4 f32  = 5888 B
constexpr int OFF_RED = OFF_M4 + 5888;           // 8 f32      = 32 B
constexpr int OFF_W2  = OFF_RED + 32;            // 32768 h    = 65536 B
constexpr int OFF_W3  = OFF_W2 + 65536;          // 8192 h     = 16384 B  (contiguous w/ W2)
constexpr int OFF_W4  = OFF_W3 + 16384;          // 256 h      = 512 B    (contiguous w/ W3)
constexpr int OFF_BIAS= OFF_W4 + 512;            // 452 f32    = 1808 B
constexpr int SMEM_BYTES = OFF_BIAS + 1808;      // 277488 B < 320 KB WGP LDS

// ---------------- helpers ----------------
__device__ __forceinline__ v8f wmma16(v16h a, v16h b, v8f c) {
  // D = A(16x32 f16) * B(32x16 f16) + C(16x16 f32)
  return __builtin_amdgcn_wmma_f32_16x16x32_f16(
      /*neg_a=*/false, a, /*neg_b=*/false, b,
      /*c_mod=*/(short)0, c, /*reuse_a=*/false, /*reuse_b=*/false);
}

// Load one lane's 16-half WMMA fragment: two contiguous 8-half (16B) runs at
// base[0..7] and base[16..23] (base already includes the lane's kb offset).
__device__ __forceinline__ v16h load_frag(const _Float16* p) {
  union { v16h h; v4f f[2]; } u;
  u.f[0] = *reinterpret_cast<const v4f*>(p);
  u.f[1] = *reinterpret_cast<const v4f*>(p + 16);
  return u.h;
}

// ---------------- kernel 0: f32 -> f16 weight conversion into d_ws ----------------
__global__ void convert_weights(const float* __restrict__ w1, const float* __restrict__ w2,
                                const float* __restrict__ w3, const float* __restrict__ w4,
                                _Float16* __restrict__ wh) {
  const int total = W1N + W2N + W3N + W4N;
  for (int i = blockIdx.x * blockDim.x + threadIdx.x; i < total;
       i += gridDim.x * blockDim.x) {
    float v;
    if      (i < W1N)             v = w1[i];
    else if (i < W1N + W2N)       v = w2[i - W1N];
    else if (i < W1N + W2N + W3N) v = w3[i - W1N - W2N];
    else                          v = w4[i - W1N - W2N - W3N];
    wh[i] = (_Float16)v;
  }
}

// ---------------- kernel 1: fully fused per-batch pipeline ----------------
// __launch_bounds__(256, 2): cap VGPRs so >=2 wave32 fit per SIMD32 — our
// 8-wave workgroup must be co-resident on one WGP (4 SIMDs x 2 waves).
__global__ __launch_bounds__(256, 2) void spec_encoder_fused(
    const float* __restrict__ x,
    const float* __restrict__ b1, const float* __restrict__ b2,
    const float* __restrict__ b3, const float* __restrict__ b4,
    const _Float16* __restrict__ w1h, const _Float16* __restrict__ w234h,
    float* __restrict__ out) {
  extern __shared__ char smem[];
  float*    s_xp  = reinterpret_cast<float*>(smem + OFF_XP);
  _Float16* s_h1  = reinterpret_cast<_Float16*>(smem + OFF_H1);
  _Float16* s_m3  = reinterpret_cast<_Float16*>(smem + OFF_M3);
  _Float16* s_h2  = reinterpret_cast<_Float16*>(smem + OFF_H2);
  float*    s_m4  = reinterpret_cast<float*>(smem + OFF_M4);
  float*    s_red = reinterpret_cast<float*>(smem + OFF_RED);
  _Float16* s_w2  = reinterpret_cast<_Float16*>(smem + OFF_W2);
  _Float16* s_w3  = reinterpret_cast<_Float16*>(smem + OFF_W3);
  _Float16* s_w4  = reinterpret_cast<_Float16*>(smem + OFF_W4);
  float*    s_bs  = reinterpret_cast<float*>(smem + OFF_BIAS);
  float*    s_b1  = s_bs;        // 256
  float*    s_b2  = s_bs + 256;  // 128
  float*    s_b3  = s_bs + 384;  // 64
  float*    s_b4  = s_bs + 448;  // 4

  const int b    = blockIdx.x;
  const int tid  = threadIdx.x;
  const int wave = tid >> 5;
  const int lane = tid & 31;
  const int nl   = lane & 15;          // N column / M row within 16-block
  const int hi16 = lane >> 4;          // 0: lanes 0-15, 1: lanes 16-31
  const int kb   = hi16 * 8;           // per-lane K offset in A/B fragments
  const int mOff = hi16 * 8;           // per-lane M offset in C/D fragments

  // ---- stage padded x into LDS: xp[0..1]=0, xp[2..301]=x, xp[302..303]=0 ----
  for (int i = tid; i < 304; i += 256) {
    float v = 0.f;
    if (i >= 2 && i < 302) v = x[b * FEAT + (i - 2)];
    s_xp[i] = v;
  }
  // ---- stage w2|w3|w4 (contiguous 41216 halfs in ws) into LDS as 16B chunks ----
  {
    const v4f* src = reinterpret_cast<const v4f*>(w234h);
    v4f*       dst = reinterpret_cast<v4f*>(s_w2);
    for (int i = tid; i < W234N / 8; i += 256) dst[i] = src[i];
  }
  // ---- stage biases ----
  for (int i = tid; i < 452; i += 256) {
    float v;
    if      (i < 256) v = b1[i];
    else if (i < 384) v = b2[i - 256];
    else if (i < 448) v = b3[i - 384];
    else              v = b4[i - 448];
    s_bs[i] = v;
  }
  __syncthreads();

  _Float16* myH1 = s_h1 + wave * 16 * H1_STRIDE;
  _Float16* myM3 = s_m3 + wave * 16 * M3_STRIDE;

  for (int tile = wave; tile < NTILE; tile += 8) {
    const int posRow = tile * 16 + nl;                 // this lane's A-matrix row
    const int pc = posRow < NPOS ? posRow : (NPOS - 1);
    const int jh = pc / 19;                            // h index
    const int lw = pc % 19;                            // w index

    // Column-side x values for this lane's A-fragment pattern: independent of ks.
    float xc[8];
    #pragma unroll
    for (int hv = 0; hv < 2; ++hv)
      #pragma unroll
      for (int q = 0; q < 4; ++q)
        xc[hv * 4 + q] = s_xp[((kb + hv * 16 + 2 * q) >> 1) * 19 + lw];

    // ================= Layer 1: 512 -> 256, leaky =================
    // A (16x512) synthesized on the fly; two N-halves of 128 to cap VGPRs.
    for (int halfN = 0; halfN < 2; ++halfN) {
      v8f acc[8] = {};
      for (int ks = 0; ks < 16; ++ks) {                // K blocks of 32
        union { v16h h; _Float16 e[16]; } A;
        const float xr = s_xp[ks * 19 + jh];           // i == ks within this K block
        #pragma unroll
        for (int t = 0; t < 8; ++t) {
          const float di = xc[t] - xr;
          const float nd = di * __builtin_amdgcn_rcpf(xc[t] + xr + EPS);
          A.e[2 * t]     = (_Float16)di;               // even klocal -> DI
          A.e[2 * t + 1] = (_Float16)nd;               // odd  klocal -> NDI
        }
        const int kcol = ks * 32 + kb;
        v16h Bf[8];
        #pragma unroll
        for (int nt = 0; nt < 8; ++nt)                 // one load clause, one wait
          Bf[nt] = load_frag(w1h + ((halfN * 8 + nt) * 16 + nl) * 512 + kcol);
        #pragma unroll
        for (int nt = 0; nt < 8; ++nt)                 // back-to-back WMMAs
          acc[nt] = wmma16(A.h, Bf[nt], acc[nt]);
      }
      #pragma unroll
      for (int nt = 0; nt < 8; ++nt) {
        const int ch = (halfN * 8 + nt) * 16 + nl;
        const float bias = s_b1[ch];
        #pragma unroll
        for (int v = 0; v < 8; ++v) {
          float val = acc[nt][v] + bias;
          val = val >= 0.f ? val : 0.01f * val;        // leaky
          myH1[(v + mOff) * H1_STRIDE + ch] = (_Float16)val;
        }
      }
    }

    // ================= Layer 2: 256 -> 128, leaky (kept as h2) =================
    {
      v8f acc[8] = {};
      for (int ks = 0; ks < 8; ++ks) {
        const int kcol = ks * 32 + kb;
        v16h A = load_frag(myH1 + nl * H1_STRIDE + kcol);
        v16h Bf[8];
        #pragma unroll
        for (int nt = 0; nt < 8; ++nt)
          Bf[nt] = load_frag(s_w2 + (nt * 16 + nl) * 256 + kcol);
        #pragma unroll
        for (int nt = 0; nt < 8; ++nt)
          acc[nt] = wmma16(A, Bf[nt], acc[nt]);
      }
      #pragma unroll
      for (int nt = 0; nt < 8; ++nt) {
        const int ch = nt * 16 + nl;
        const float bias = s_b2[ch];
        #pragma unroll
        for (int v = 0; v < 8; ++v) {
          float val = acc[nt][v] + bias;
          val = val >= 0.f ? val : 0.01f * val;
          const int pg = tile * 16 + v + mOff;         // global position row (< 368)
          s_h2[pg * H2_STRIDE + ch] = (_Float16)val;
        }
      }
    }

    // ================= Layer 3: 128 -> 64, relu =================
    {
      v8f acc[4] = {};
      #pragma unroll
      for (int ks = 0; ks < 4; ++ks) {
        const int kcol = ks * 32 + kb;
        v16h A = load_frag(s_h2 + (tile * 16 + nl) * H2_STRIDE + kcol);
        v16h Bf[4];
        #pragma unroll
        for (int nt = 0; nt < 4; ++nt)
          Bf[nt] = load_frag(s_w3 + (nt * 16 + nl) * 128 + kcol);
        #pragma unroll
        for (int nt = 0; nt < 4; ++nt)
          acc[nt] = wmma16(A, Bf[nt], acc[nt]);
      }
      #pragma unroll
      for (int nt = 0; nt < 4; ++nt) {
        const int ch = nt * 16 + nl;
        const float bias = s_b3[ch];
        #pragma unroll
        for (int v = 0; v < 8; ++v) {
          float val = fmaxf(acc[nt][v] + bias, 0.f);
          myM3[(v + mOff) * M3_STRIDE + ch] = (_Float16)val;
        }
      }
    }

    // ================= Layer 4: 64 -> 4, relu (N padded to 16) =================
    {
      v8f acc = {};
      #pragma unroll
      for (int ks = 0; ks < 2; ++ks) {
        const int kcol = ks * 32 + kb;
        v16h A = load_frag(myM3 + nl * M3_STRIDE + kcol);
        union { v16h h; _Float16 e[16]; } B;
        if (nl < 4) {
          B.h = load_frag(s_w4 + nl * 64 + kcol);
        } else {
          #pragma unroll
          for (int e = 0; e < 16; ++e) B.e[e] = (_Float16)0.f;
        }
        acc = wmma16(A, B.h, acc);
      }
      if (nl < 4) {
        const float bias = s_b4[nl];
        #pragma unroll
        for (int v = 0; v < 8; ++v) {
          const float val = fmaxf(acc[v] + bias, 0.f);
          const int pg = tile * 16 + v + mOff;         // pg < 368; rows >= 361 ignored
          s_m4[pg * 4 + nl] = val;
        }
      }
    }
  }

  __syncthreads();

  // ---- softmax stats over 361 positions for each of the 4 channel groups ----
  if (tid < 4) {
    float mx = -3.0e38f;
    for (int p = 0; p < NPOS; ++p) mx = fmaxf(mx, s_m4[p * 4 + tid]);
    float sm = 0.f;
    for (int p = 0; p < NPOS; ++p) sm += __expf(s_m4[p * 4 + tid] - mx);
    s_red[tid]     = mx;
    s_red[4 + tid] = 1.f / sm;
  }
  __syncthreads();

  // ---- logits -> probabilities in place ----
  for (int p = tid; p < NPOS; p += 256) {
    #pragma unroll
    for (int g = 0; g < 4; ++g) {
      s_m4[p * 4 + g] = __expf(s_m4[p * 4 + g] - s_red[g]) * s_red[4 + g];
    }
  }
  __syncthreads();

  // ---- probability-weighted sum of h2 -> out[b, 128] ----
  if (tid < 128) {
    const int g = tid >> 5;                            // channel group = ch / 32
    float acc = 0.f;
    for (int p = 0; p < NPOS; ++p) {
      acc += (float)s_h2[p * H2_STRIDE + tid] * s_m4[p * 4 + g];
    }
    out[b * 128 + tid] = acc;
  }
}

// ---------------- host-side launcher ----------------
extern "C" void kernel_launch(void* const* d_in, const int* in_sizes, int n_in,
                              void* d_out, int out_size, void* d_ws, size_t ws_size,
                              hipStream_t stream) {
  const float* x  = (const float*)d_in[0];
  const float* w1 = (const float*)d_in[1];
  const float* b1 = (const float*)d_in[2];
  const float* w2 = (const float*)d_in[3];
  const float* b2 = (const float*)d_in[4];
  const float* w3 = (const float*)d_in[5];
  const float* b3 = (const float*)d_in[6];
  const float* w4 = (const float*)d_in[7];
  const float* b4 = (const float*)d_in[8];
  float* out = (float*)d_out;

  _Float16* wh = (_Float16*)d_ws;              // needs 344,576 B of workspace
  _Float16* w1h   = wh;
  _Float16* w234h = wh + W1N;                  // w2|w3|w4 contiguous

  const int convTotal = W1N + W2N + W3N + W4N;
  convert_weights<<<(convTotal + 255) / 256, 256, 0, stream>>>(w1, w2, w3, w4, wh);

  spec_encoder_fused<<<BATCH, 256, SMEM_BYTES, stream>>>(
      x, b1, b2, b3, b4, w1h, w234h, out);
}
